// Attention_68015102099893
// MI455X (gfx1250) — matrix operand order repaired and checked by reading, compiled, not run on hardware
//
#include <hip/hip_runtime.h>
#include <hip/hip_bf16.h>
#include <math.h>

typedef __attribute__((ext_vector_type(16))) __bf16 v16bf;
typedef __attribute__((ext_vector_type(8)))  float  v8f;

#define C_DIM   512
#define HID     256
#define ROWS    81     // N*P tokens per image
#define ROWS_PAD 96    // padded to 6 M-tiles of 16
#define KSTEPS  16     // 512 / 32
#define NTILES  16     // 256 / 16
#define MTILES  6      // 96 / 16

__device__ __forceinline__ unsigned short f2bf(float f) {
    unsigned int u = __float_as_uint(f);
    u += 0x7FFFu + ((u >> 16) & 1u);   // round-to-nearest-even
    return (unsigned short)(u >> 16);
}
__device__ __forceinline__ float bf2f(unsigned short h) {
    return __uint_as_float(((unsigned int)h) << 16);
}

// ---------------------------------------------------------------------------
// Kernel 1: pack W_fc [256,512] f32 into bf16 B-fragment layout:
//   wp[((ks*16 + nt)*32 + lane)*16 + pos], K=ks*32 + (lane>>4)*8 + (pos&7) + (pos>>3)*16
//   column h = nt*16 + (lane&15)
// ---------------------------------------------------------------------------
__global__ void pack_w_kernel(const float* __restrict__ W_fc,
                              unsigned short* __restrict__ wp) {
    int idx  = blockIdx.x * 256 + threadIdx.x;   // 0 .. 131071
    int pos  = idx & 15;
    int lane = (idx >> 4) & 31;
    int nt   = (idx >> 9) & 15;
    int ks   = idx >> 13;
    int k = ks * 32 + ((lane >> 4) << 3) + (pos & 7) + ((pos >> 3) << 4);
    int h = nt * 16 + (lane & 15);
    wp[idx] = f2bf(W_fc[h * C_DIM + k]);
}

// ---------------------------------------------------------------------------
// Kernel 2: one block (256 threads = 8 waves) per image.
// ---------------------------------------------------------------------------
__global__ void __launch_bounds__(256)
attn_kernel(const float* __restrict__ x,
            const unsigned short* __restrict__ wp,
            const float* __restrict__ b_fc,
            float* __restrict__ out) {
    extern __shared__ __align__(128) char smem[];
    unsigned short* xp = (unsigned short*)smem;               // [96 rows][1024 bf16] A-frag packed (96 KB)
    float* xq = (float*)(smem + ROWS_PAD * C_DIM * 2);        // [96][256] f32 (96 KB)
    __shared__ float s_attn[81];
    __shared__ float s_attn2[81];
    __shared__ float s_part[243];

    const int tid = threadIdx.x;
    const int b   = blockIdx.x;
    const float* xb = x + (size_t)b * (ROWS * C_DIM);

    // ---- Stage A: load x, convert to bf16, scatter into A-fragment layout ----
    // packed byte offset for (row, c): row*2048 + ((kk*2 + half)<<5) + pos*2
    //   kk = c/32, g = (c/8)%4, half = g&1, pos = (g>>1)*8 + c%8
    for (int idx = tid; idx < ROWS * (C_DIM / 4); idx += 256) {
        int q  = idx >> 7;            // pixel 0..80 (h*9+w)
        int c4 = idx & 127;
        int c  = c4 << 2;
        int h = q / 9, w = q % 9;
        int n = (h / 3) * 3 + (w / 3);
        int p = (h % 3) * 3 + (w % 3);
        int row = n * 9 + p;
        float4 v = *(const float4*)(xb + q * C_DIM + c);
        unsigned int lo = (unsigned int)f2bf(v.x) | ((unsigned int)f2bf(v.y) << 16);
        unsigned int hi = (unsigned int)f2bf(v.z) | ((unsigned int)f2bf(v.w) << 16);
        int kk   = c >> 5;
        int g    = (c >> 3) & 3;
        int half = g & 1;
        int pos  = ((g >> 1) << 3) + (c & 7);
        int off  = row * 2048 + (((kk << 1) + half) << 5) + (pos << 1);
        *(uint2*)((char*)xp + off) = make_uint2(lo, hi);
    }
    // zero padding rows 81..95 (15 rows * 2048 B = 3840 uint2)
    for (int idx = tid; idx < 3840; idx += 256) {
        *(uint2*)((char*)xp + ROWS * 2048 + idx * 8) = make_uint2(0u, 0u);
    }
    __syncthreads();

    // ---- Stage B: GEMM xq = x_bf16 @ W^T + b, via bf16 WMMA ----
    const int wave = tid >> 5;
    const int lane = tid & 31;
    const int nt0  = wave * 2;

    v8f acc[MTILES][2];
#pragma unroll
    for (int mt = 0; mt < MTILES; ++mt) { acc[mt][0] = {}; acc[mt][1] = {}; }

    const v16bf* wpv = (const v16bf*)wp;
    const int arow  = lane & 15;
    const int ahalf = (lane >> 4) & 1;

#pragma unroll 2
    for (int ks = 0; ks < KSTEPS; ++ks) {
        v16bf bf0 = wpv[(ks * NTILES + nt0)     * 32 + lane];
        v16bf bf1 = wpv[(ks * NTILES + nt0 + 1) * 32 + lane];
#pragma unroll
        for (int mt = 0; mt < MTILES; ++mt) {
            int row = mt * 16 + arow;
            const v16bf* ap =
                (const v16bf*)((char*)xp + row * 2048 + (((ks << 1) + ahalf) << 5));
            v16bf af = *ap;
            acc[mt][0] = __builtin_amdgcn_wmma_f32_16x16x32_bf16(
                false, af, false, bf0, (short)0, acc[mt][0], false, false);
            acc[mt][1] = __builtin_amdgcn_wmma_f32_16x16x32_bf16(
                false, af, false, bf1, (short)0, acc[mt][1], false, false);
        }
    }

    // bias + store xq tiles to LDS (C/D layout: VGPR j -> M = mt*16 + j + (lane>=16)*8)
    float bias0 = b_fc[nt0 * 16 + (lane & 15)];
    float bias1 = b_fc[(nt0 + 1) * 16 + (lane & 15)];
    {
        int rbase = ((lane >> 4) << 3);
        int col0  = nt0 * 16 + (lane & 15);
#pragma unroll
        for (int mt = 0; mt < MTILES; ++mt) {
#pragma unroll
            for (int j = 0; j < 8; ++j) {
                int row = mt * 16 + j + rbase;
                xq[row * HID + col0]      = acc[mt][0][j] + bias0;
                xq[row * HID + col0 + 16] = acc[mt][1][j] + bias1;
            }
        }
    }
    __syncthreads();

    // ---- Stage C: attn[n][m] = (1/48) * <xq[n], xq[m]>  (d = 9 tokens x 256) ----
    if (tid < 243) {
        int pair = tid % 81;          // n*9 + m
        int part = tid / 81;          // covers p = part*3 .. part*3+2
        int n = pair / 9, m = pair % 9;
        float dot = 0.f;
        for (int pp = 0; pp < 3; ++pp) {
            int p = part * 3 + pp;
            const float4* qa = (const float4*)(xq + (n * 9 + p) * HID);
            const float4* qb = (const float4*)(xq + (m * 9 + p) * HID);
#pragma unroll 4
            for (int i = 0; i < HID / 4; ++i) {
                float4 a = qa[i], bb = qb[i];
                dot += a.x * bb.x + a.y * bb.y + a.z * bb.z + a.w * bb.w;
            }
        }
        s_part[tid] = dot;
    }
    __syncthreads();
    if (tid < 81) {
        int n = tid / 9, m = tid % 9;
        float dot = s_part[tid] + s_part[tid + 81] + s_part[tid + 162];
        s_attn[tid] = dot * (1.0f / 48.0f) + ((n == m) ? -100.0f : 0.0f);
    }
    __syncthreads();

    // ---- Stage D: attn2 = softmax( (attn @ attn^T) / 3 ) ----
    if (tid < 81) {
        int n = tid / 9, m = tid % 9;
        float s = 0.f;
#pragma unroll
        for (int k = 0; k < 9; ++k) s += s_attn[n * 9 + k] * s_attn[m * 9 + k];
        s_attn2[tid] = s * (1.0f / 3.0f);
    }
    __syncthreads();
    if (tid < 9) {
        float mx = -1e30f;
        for (int k = 0; k < 9; ++k) mx = fmaxf(mx, s_attn2[tid * 9 + k]);
        float e[9], sum = 0.f;
        for (int k = 0; k < 9; ++k) { e[k] = __expf(s_attn2[tid * 9 + k] - mx); sum += e[k]; }
        float inv = 1.f / sum;
        for (int k = 0; k < 9; ++k) s_attn2[tid * 9 + k] = e[k] * inv;
    }
    __syncthreads();

    // ---- Stage E: attn = softmax(attn + attn2) ----
    if (tid < 81) s_attn[tid] += s_attn2[tid];
    __syncthreads();
    if (tid < 9) {
        float mx = -1e30f;
        for (int k = 0; k < 9; ++k) mx = fmaxf(mx, s_attn[tid * 9 + k]);
        float e[9], sum = 0.f;
        for (int k = 0; k < 9; ++k) { e[k] = __expf(s_attn[tid * 9 + k] - mx); sum += e[k]; }
        float inv = 1.f / sum;
        for (int k = 0; k < 9; ++k) s_attn[tid * 9 + k] = e[k] * inv;
    }
    __syncthreads();

    // ---- Stage H: out = attn @ v, v read back from bf16 LDS copy ----
    float* outb = out + (size_t)b * (ROWS * C_DIM);
    for (int idx = tid; idx < ROWS * (C_DIM / 4); idx += 256) {
        int q  = idx >> 7;
        int c4 = idx & 127;
        int c  = c4 << 2;
        int h = q / 9, w = q % 9;
        int n = (h / 3) * 3 + (w / 3);
        int p = (h % 3) * 3 + (w % 3);
        int kk   = c >> 5;
        int g    = (c >> 3) & 3;
        int half = g & 1;
        int pos  = ((g >> 1) << 3) + (c & 7);
        int coff = (((kk << 1) + half) << 5) + (pos << 1);
        float4 accv = {0.f, 0.f, 0.f, 0.f};
#pragma unroll
        for (int m = 0; m < 9; ++m) {
            float a = s_attn[n * 9 + m];
            int rowv = m * 9 + p;
            uint2 pk = *(const uint2*)((const char*)xp + rowv * 2048 + coff);
            accv.x += a * bf2f((unsigned short)(pk.x & 0xFFFFu));
            accv.y += a * bf2f((unsigned short)(pk.x >> 16));
            accv.z += a * bf2f((unsigned short)(pk.y & 0xFFFFu));
            accv.w += a * bf2f((unsigned short)(pk.y >> 16));
        }
        *(float4*)(outb + q * C_DIM + c) = accv;
    }
}

extern "C" void kernel_launch(void* const* d_in, const int* in_sizes, int n_in,
                              void* d_out, int out_size, void* d_ws, size_t ws_size,
                              hipStream_t stream) {
    const float* x    = (const float*)d_in[0];
    const float* W_fc = (const float*)d_in[1];
    const float* b_fc = (const float*)d_in[2];
    float* out = (float*)d_out;
    unsigned short* wp = (unsigned short*)d_ws;   // 512*256 bf16 = 256 KB

    int B = in_sizes[0] / (ROWS * C_DIM);         // 2048

    pack_w_kernel<<<(C_DIM * HID) / 256, 256, 0, stream>>>(W_fc, wp);

    size_t smem_bytes = (size_t)ROWS_PAD * C_DIM * 2   // bf16 x, packed   (96 KB)
                      + (size_t)ROWS_PAD * HID * 4;    // f32 xq           (96 KB)
    attn_kernel<<<B, 256, smem_bytes, stream>>>(x, wp, b_fc, out);
}
